// Qwen35Layer_13039520711500
// MI455X (gfx1250) — matrix operand order repaired and checked by reading, not compiled, not run on hardware
//
#include <hip/hip_runtime.h>
#include <hip/hip_bf16.h>

// ---------------------------------------------------------------------------
// Types / helpers for CDNA5 WMMA (wave32, v_wmma_f32_16x16x32_bf16)
// ---------------------------------------------------------------------------
typedef __attribute__((ext_vector_type(16))) __bf16 v16bf;
typedef __attribute__((ext_vector_type(8)))  float  v8f;

struct __align__(16) B16v { unsigned int w[4]; };
union FragU { v16bf v; B16v b[2]; };

#ifndef __has_builtin
#define __has_builtin(x) 0
#endif
#if __has_builtin(__builtin_amdgcn_global_load_async_to_lds_b128) && \
    __has_builtin(__builtin_amdgcn_s_wait_asynccnt)
#define ASYNC_LDS 1
#else
#define ASYNC_LDS 0
#endif

#if ASYNC_LDS
typedef int v4i32 __attribute__((vector_size(16)));
#endif

// 16-byte global -> LDS copy. Async (ASYNCcnt-tracked, no VGPR round trip) when
// the gfx1250 builtin exists; plain load+ds_store otherwise.
__device__ __forceinline__ void cp16_g2l(const void* gsrc, void* ldst) {
#if ASYNC_LDS
    __builtin_amdgcn_global_load_async_to_lds_b128(
        (__attribute__((address_space(1))) v4i32*)(unsigned long long)(size_t)gsrc,
        (__attribute__((address_space(3))) v4i32*)(unsigned)(size_t)ldst,
        0, 0);
#else
    *reinterpret_cast<B16v*>(ldst) = *reinterpret_cast<const B16v*>(gsrc);
#endif
}

template <int N>
__device__ __forceinline__ void async_wait() {
#if ASYNC_LDS
    __builtin_amdgcn_s_wait_asynccnt((unsigned short)N);
#endif
}

__device__ __forceinline__ unsigned short f2bf(float f) {
    unsigned int u = __float_as_uint(f);
    u += 0x7FFFu + ((u >> 16) & 1u);           // round-to-nearest-even
    return (unsigned short)(u >> 16);
}

__device__ __forceinline__ v8f zero8() {
    v8f z;
#pragma unroll
    for (int i = 0; i < 8; ++i) z[i] = 0.f;
    return z;
}

// A-operand fragment (16xK=32 tile, row-major source, stride in ushorts).
// Lane l<16: row=l, K={0..7,16..23}; lane>=16: row=l-16, K={8..15,24..31}.
__device__ __forceinline__ v16bf ld_fragA(const unsigned short* p, int stride,
                                          int row_base, int k_base, int lane) {
    const int lm = lane & 15;
    const int hi = (lane >> 4) & 1;
    const unsigned short* q = p + (size_t)(row_base + lm) * stride + k_base + hi * 8;
    FragU f;
    f.b[0] = *reinterpret_cast<const B16v*>(q);
    f.b[1] = *reinterpret_cast<const B16v*>(q + 16);
    return f.v;
}

// B-operand fragment from B^T stored row-major [N, K].
// Lane l<16: n=l, K={0..7,16..23}; lane>=16: n=l-16, K={8..15,24..31}.
__device__ __forceinline__ v16bf ld_fragB(const unsigned short* p, int stride,
                                          int row_base, int k_base, int lane) {
    const int lm = lane & 15;
    const int hi = (lane >> 4) & 1;
    const unsigned short* q = p + (size_t)(row_base + lm) * stride + k_base + hi * 8;
    FragU f;
    f.b[0] = *reinterpret_cast<const B16v*>(q);
    f.b[1] = *reinterpret_cast<const B16v*>(q + 16);
    return f.v;
}

__device__ __forceinline__ v8f wmma_bf16(v16bf a, v16bf b, v8f c) {
    return __builtin_amdgcn_wmma_f32_16x16x32_bf16(false, a, false, b, (short)0, c,
                                                   false, false);
}

// ---------------------------------------------------------------------------
// LayerNorm: one block per token row (D=1024), writes f32 + bf16
// ---------------------------------------------------------------------------
__global__ __launch_bounds__(256) void k_layernorm(const float* __restrict__ x,
                                                   const float* __restrict__ g,
                                                   const float* __restrict__ bta,
                                                   float* __restrict__ yf,
                                                   unsigned short* __restrict__ yb) {
    const int row = blockIdx.x;
    const int tid = threadIdx.x;
    const float* xr = x + (size_t)row * 1024;
    __shared__ float red[256];

    float s = 0.f;
    for (int d = tid; d < 1024; d += 256) s += xr[d];
    red[tid] = s;
    __syncthreads();
    for (int off = 128; off > 0; off >>= 1) {
        if (tid < off) red[tid] += red[tid + off];
        __syncthreads();
    }
    const float mu = red[0] * (1.f / 1024.f);
    __syncthreads();

    float v = 0.f;
    for (int d = tid; d < 1024; d += 256) {
        float t = xr[d] - mu;
        v += t * t;
    }
    red[tid] = v;
    __syncthreads();
    for (int off = 128; off > 0; off >>= 1) {
        if (tid < off) red[tid] += red[tid + off];
        __syncthreads();
    }
    const float rstd = rsqrtf(red[0] * (1.f / 1024.f) + 1e-5f);

    for (int d = tid; d < 1024; d += 256) {
        float t = (xr[d] - mu) * rstd * g[d] + bta[d];
        yf[(size_t)row * 1024 + d] = t;
        yb[(size_t)row * 1024 + d] = f2bf(t);
    }
}

// ---------------------------------------------------------------------------
// f32 -> bf16 convert (flat)
// ---------------------------------------------------------------------------
__global__ __launch_bounds__(256) void k_cvt(const float* __restrict__ in,
                                             unsigned short* __restrict__ out,
                                             size_t n) {
    size_t i = (size_t)blockIdx.x * 256 + threadIdx.x;
    if (i < n) out[i] = f2bf(in[i]);
}

// f32 [K,N] -> bf16 [N,K] (transpose-convert; B^T staging for WMMA B operand)
__global__ __launch_bounds__(256) void k_tcvt(const float* __restrict__ in,
                                              unsigned short* __restrict__ out,
                                              int K, int N) {
    size_t idx = (size_t)blockIdx.x * 256 + threadIdx.x;
    if (idx >= (size_t)K * N) return;
    int k = (int)(idx / N);
    int n = (int)(idx % N);
    out[(size_t)n * K + k] = f2bf(in[idx]);
}

// ---------------------------------------------------------------------------
// bf16 WMMA GEMM: C[M,N] = A[M,K] * B[K,N]  (B passed as B^T [N,K])
// Block tile 128x128, 8 waves (4x2), wave tile 32x64 (2x4 WMMA), K-step 32.
// Double-buffered LDS staged with async global->LDS copies (ASYNCcnt) when
// available. mode 0: store f32   mode 1: gelu->bf16   mode 2: Cf += scale[m]*v
// ---------------------------------------------------------------------------
__global__ __launch_bounds__(256) void k_gemm(const unsigned short* __restrict__ A,
                                              const unsigned short* __restrict__ Bt,
                                              float* __restrict__ Cf,
                                              unsigned short* __restrict__ Cb,
                                              const float* __restrict__ scale,
                                              int sstride, int scol,
                                              int N, int K, int mode) {
    __shared__ unsigned short Ash[2][128 * 32];
    __shared__ unsigned short Bsh[2][128 * 32];
    const int tid  = threadIdx.x;
    const int lane = tid & 31, wave = tid >> 5;
    const int wm = wave >> 1, wn = wave & 1;
    const int lm = lane & 15, hi = (lane >> 4) & 1;
    const size_t am0 = (size_t)blockIdx.y * 128;
    const size_t bn0 = (size_t)blockIdx.x * 128;

    v8f acc[2][4];
#pragma unroll
    for (int i = 0; i < 2; ++i)
#pragma unroll
        for (int j = 0; j < 4; ++j) acc[i][j] = zero8();

    const int nk = K >> 5;

    // Stage one 128x32 A tile + 128x32 B^T tile into buffer `buf`.
    // 512 16B chunks per tile; 256 threads x 2 chunks; 4 async ops per wave.
    auto stage = [&](int buf, int kt) {
#pragma unroll
        for (int j = 0; j < 2; ++j) {
            int id = tid * 2 + j;
            int r = id >> 2, ch = id & 3;
            cp16_g2l(A + (am0 + r) * K + kt * 32 + ch * 8,
                     &Ash[buf][r * 32 + ch * 8]);
            cp16_g2l(Bt + (bn0 + r) * K + kt * 32 + ch * 8,
                     &Bsh[buf][r * 32 + ch * 8]);
        }
    };

    stage(0, 0);
    for (int kt = 0; kt < nk; ++kt) {
        if (kt + 1 < nk) {
            stage((kt + 1) & 1, kt + 1);
            async_wait<4>();   // async loads retire in order: tile kt is resident
        } else {
            async_wait<0>();
        }
        __syncthreads();

        const unsigned short* Ab = Ash[kt & 1];
        const unsigned short* Bb = Bsh[kt & 1];
        v16bf a[2], bb[4];
#pragma unroll
        for (int i = 0; i < 2; ++i) a[i] = ld_fragA(Ab, 32, wm * 32 + i * 16, 0, lane);
#pragma unroll
        for (int j = 0; j < 4; ++j) bb[j] = ld_fragB(Bb, 32, wn * 64 + j * 16, 0, lane);
#pragma unroll
        for (int i = 0; i < 2; ++i)
#pragma unroll
            for (int j = 0; j < 4; ++j) acc[i][j] = wmma_bf16(a[i], bb[j], acc[i][j]);
        __syncthreads();   // all waves done reading before buffer is re-staged
    }

#pragma unroll
    for (int i = 0; i < 2; ++i)
#pragma unroll
        for (int j = 0; j < 4; ++j)
#pragma unroll
            for (int r = 0; r < 8; ++r) {
                size_t m = am0 + wm * 32 + i * 16 + r + hi * 8;
                size_t n = bn0 + wn * 64 + j * 16 + lm;
                float v = acc[i][j][r];
                if (mode == 0) {
                    Cf[m * N + n] = v;
                } else if (mode == 1) {
                    float gx = 0.5f * v * (1.f + erff(v * 0.70710678f));
                    Cb[m * N + n] = f2bf(gx);
                } else {
                    Cf[m * N + n] += scale[m * sstride + scol] * v;
                }
            }
}

// ---------------------------------------------------------------------------
// Flash attention (dense branch) with WMMA for QK^T and PV.
// grid (S/64, H, B), block 128 (4 waves); wave owns 16 q rows; 64-key chunks.
// ---------------------------------------------------------------------------
__global__ __launch_bounds__(128) void k_flash(const unsigned short* __restrict__ Qb,
                                               const unsigned short* __restrict__ Kb,
                                               const unsigned short* __restrict__ Vb,
                                               const int* __restrict__ mask,
                                               float* __restrict__ Of) {
    const int S = 1024, Dm = 1024;
    const int qblk = blockIdx.x, h = blockIdx.y, b = blockIdx.z;
    const int tid = threadIdx.x;
    const int wave = tid >> 5, lane = tid & 31;
    const int lm = lane & 15, hi = (lane >> 4) & 1;

    __shared__ unsigned short Ksh[64 * 64];      // K chunk, row-major [key, dk]
    __shared__ unsigned short Vsh[64 * 64];      // V^T chunk [dk, key]
    __shared__ unsigned short Psh[4 * 16 * 64];  // per-wave P tile

    const int rowbase = b * S + qblk * 64 + wave * 16;  // global token row
    const int colbase = h * 64;

    v16bf aq[2];
#pragma unroll
    for (int kh = 0; kh < 2; ++kh)
        aq[kh] = ld_fragA(Qb, Dm, rowbase, colbase + kh * 32, lane);

    v8f o[4];
    float mi[8], li[8];
#pragma unroll
    for (int nt = 0; nt < 4; ++nt) o[nt] = zero8();
#pragma unroll
    for (int r = 0; r < 8; ++r) { mi[r] = -3.0e38f; li[r] = 0.f; }

    for (int kb = 0; kb < S / 64; ++kb) {
        // stage K chunk (512 x 16B, 4 per thread)
#pragma unroll
        for (int j = 0; j < 4; ++j) {
            int id = tid * 4 + j;
            int r = id >> 3, ch = id & 7;
            const B16v* src = reinterpret_cast<const B16v*>(
                Kb + (size_t)(b * S + kb * 64 + r) * Dm + colbase + ch * 8);
            *reinterpret_cast<B16v*>(Ksh + r * 64 + ch * 8) = *src;
            __builtin_prefetch(src + (size_t)8 * Dm / 8, 0, 1);  // 8 key rows ahead
        }
        // stage V chunk transposed
#pragma unroll
        for (int j = 0; j < 4; ++j) {
            int task = tid + j * 128;
            int k = task >> 3, dc = (task & 7) * 8;
            B16v val = *reinterpret_cast<const B16v*>(
                Vb + (size_t)(b * S + kb * 64 + k) * Dm + colbase + dc);
            const unsigned short* vv = reinterpret_cast<const unsigned short*>(&val);
#pragma unroll
            for (int e = 0; e < 8; ++e) Vsh[(dc + e) * 64 + k] = vv[e];
        }
        __syncthreads();

        // scores: 16x64 per wave (4 N tiles, K=64 over dk -> 2 WMMA each)
        float sv[4][8];
#pragma unroll
        for (int nt = 0; nt < 4; ++nt) {
            v8f c = zero8();
#pragma unroll
            for (int kh = 0; kh < 2; ++kh) {
                v16bf bk = ld_fragB(Ksh, 64, nt * 16, kh * 32, lane);
                c = wmma_bf16(aq[kh], bk, c);
            }
#pragma unroll
            for (int r = 0; r < 8; ++r) {
                int q = qblk * 64 + wave * 16 + r + hi * 8;
                int k = kb * 64 + nt * 16 + lm;
                float v = c[r] * 0.125f;  // 1/sqrt(64)
                if (mask[(size_t)(b * S + q) * S + k] == 0) v = -1e9f;
                sv[nt][r] = v;
            }
        }
        // online softmax: row stats across the 16 N-lanes
        float psc[8];
#pragma unroll
        for (int r = 0; r < 8; ++r) {
            float v = sv[0][r];
#pragma unroll
            for (int nt = 1; nt < 4; ++nt) v = fmaxf(v, sv[nt][r]);
#pragma unroll
            for (int mk = 1; mk < 16; mk <<= 1) v = fmaxf(v, __shfl_xor(v, mk, 32));
            float mn = fmaxf(mi[r], v);
            psc[r] = __expf(mi[r] - mn);
            mi[r] = mn;
        }
        float rs[8];
#pragma unroll
        for (int r = 0; r < 8; ++r) rs[r] = 0.f;
#pragma unroll
        for (int nt = 0; nt < 4; ++nt)
#pragma unroll
            for (int r = 0; r < 8; ++r) {
                float p = __expf(sv[nt][r] - mi[r]);
                rs[r] += p;
                Psh[wave * 1024 + (r + hi * 8) * 64 + nt * 16 + lm] = f2bf(p);
            }
#pragma unroll
        for (int r = 0; r < 8; ++r) {
            float v = rs[r];
#pragma unroll
            for (int mk = 1; mk < 16; mk <<= 1) v += __shfl_xor(v, mk, 32);
            li[r] = li[r] * psc[r] + v;
        }
#pragma unroll
        for (int nt = 0; nt < 4; ++nt)
#pragma unroll
            for (int r = 0; r < 8; ++r) o[nt][r] *= psc[r];
        __syncthreads();

        // PV: o += P (A-layout from LDS) * V (B operand = V^T rows)
        v16bf pa[2];
#pragma unroll
        for (int kh = 0; kh < 2; ++kh)
            pa[kh] = ld_fragA(Psh + wave * 1024, 64, 0, kh * 32, lane);
#pragma unroll
        for (int nt = 0; nt < 4; ++nt)
#pragma unroll
            for (int kh = 0; kh < 2; ++kh) {
                v16bf bv = ld_fragB(Vsh, 64, nt * 16, kh * 32, lane);
                o[nt] = wmma_bf16(pa[kh], bv, o[nt]);
            }
        __syncthreads();
    }

#pragma unroll
    for (int nt = 0; nt < 4; ++nt)
#pragma unroll
        for (int r = 0; r < 8; ++r) {
            int q = rowbase + r + hi * 8;
            int d = colbase + nt * 16 + lm;
            Of[(size_t)q * Dm + d] = o[nt][r] / li[r];
        }
}

// ---------------------------------------------------------------------------
// Linear delta-attention branch (O(n), VALU; ~1 GFLOP total)
// ---------------------------------------------------------------------------
__global__ __launch_bounds__(256) void k_linctx(const float* __restrict__ Kd,
                                                const float* __restrict__ Vd,
                                                float* __restrict__ ctx,
                                                float* __restrict__ ksum) {
    const int h = blockIdx.x, b = blockIdx.y;
    const int tid = threadIdx.x;
    __shared__ float ks[16][64];
    __shared__ float vs[16][64];
    float acc[16];
#pragma unroll
    for (int i = 0; i < 16; ++i) acc[i] = 0.f;
    float ka = 0.f;

    for (int sc = 0; sc < 1024; sc += 16) {
#pragma unroll
        for (int i = 0; i < 4; ++i) {
            int idx = tid + i * 256;
            int ss = idx >> 6, d = idx & 63;
            size_t g = (size_t)(b * 1024 + sc + ss) * 1024 + h * 64 + d;
            ks[ss][d] = Kd[g];
            vs[ss][d] = Vd[g];
        }
        __syncthreads();
#pragma unroll
        for (int i = 0; i < 16; ++i) {
            int idx = tid + i * 256;
            int d = idx >> 6, e = idx & 63;
            float a = 0.f;
#pragma unroll
            for (int ss = 0; ss < 16; ++ss) a += ks[ss][d] * vs[ss][e];  // raw kd
            acc[i] += a;
        }
        if (tid < 64) {
#pragma unroll
            for (int ss = 0; ss < 16; ++ss) ka += fmaxf(ks[ss][tid], 0.f);  // relu(kd)
        }
        __syncthreads();
    }
    size_t base = (size_t)(b * 16 + h) * 4096;
#pragma unroll
    for (int i = 0; i < 16; ++i) ctx[base + tid + i * 256] = acc[i];
    if (tid < 64) ksum[(b * 16 + h) * 64 + tid] = ka;
}

__global__ __launch_bounds__(64) void k_linout(const float* __restrict__ Qd,
                                               const float* __restrict__ ctx,
                                               const float* __restrict__ ksum,
                                               float* __restrict__ dpre) {
    const int tok = blockIdx.x, h = blockIdx.y;
    const int b = tok >> 10;
    const int tid = threadIdx.x;
    __shared__ float qs[64];
    qs[tid] = fmaxf(Qd[(size_t)tok * 1024 + h * 64 + tid], 0.f);
    __syncthreads();
    const float* cx = ctx + (size_t)(b * 16 + h) * 4096;
    const float* km = ksum + (b * 16 + h) * 64;
    float out = 0.f, z = 0.f;
#pragma unroll 8
    for (int d = 0; d < 64; ++d) {
        out += qs[d] * cx[d * 64 + tid];
        z += qs[d] * km[d];
    }
    dpre[(size_t)tok * 1024 + h * 64 + tid] = out / (z + 1e-6f);
}

// ---------------------------------------------------------------------------
// Gate, combine, router, copy
// ---------------------------------------------------------------------------
__global__ __launch_bounds__(256) void k_gate(const float* __restrict__ xn1,
                                              const float* __restrict__ wg,
                                              float* __restrict__ g0, int NT) {
    int t = blockIdx.x * 256 + threadIdx.x;
    if (t >= NT) return;
    const float* xr = xn1 + (size_t)t * 1024;
    float s0 = 0.f, s1 = 0.f;
    for (int d = 0; d < 1024; ++d) {
        float v = xr[d];
        s0 += v * wg[2 * d];
        s1 += v * wg[2 * d + 1];
    }
    float m = fmaxf(s0, s1);
    float e0 = __expf(s0 - m), e1 = __expf(s1 - m);
    g0[t] = e0 / (e0 + e1);
}

__global__ __launch_bounds__(256) void k_combine(const float* __restrict__ x,
                                                 const float* __restrict__ dense,
                                                 const float* __restrict__ delta,
                                                 const float* __restrict__ g0,
                                                 float* __restrict__ x2,
                                                 int D, size_t total) {
    size_t i = (size_t)blockIdx.x * 256 + threadIdx.x;
    if (i >= total) return;
    int t = (int)(i / D);
    float g = g0[t];
    x2[i] = x[i] + g * dense[i] + (1.f - g) * delta[i];
}

__global__ __launch_bounds__(256) void k_router(const float* __restrict__ xn2,
                                                const float* __restrict__ wr,
                                                float* __restrict__ comb, int NT) {
    int t = blockIdx.x * 256 + threadIdx.x;
    if (t >= NT) return;
    float lg[8];
#pragma unroll
    for (int e = 0; e < 8; ++e) lg[e] = 0.f;
    const float* xr = xn2 + (size_t)t * 1024;
    for (int d = 0; d < 1024; ++d) {
        float v = xr[d];
#pragma unroll
        for (int e = 0; e < 8; ++e) lg[e] += v * wr[d * 8 + e];
    }
    int i1 = 0;
#pragma unroll
    for (int e = 1; e < 8; ++e) if (lg[e] > lg[i1]) i1 = e;
    int i2 = (i1 == 0) ? 1 : 0;
#pragma unroll
    for (int e = 0; e < 8; ++e) if (e != i1 && lg[e] > lg[i2]) i2 = e;
    float e2 = __expf(lg[i2] - lg[i1]);
    float inv = 1.f / (1.f + e2);
#pragma unroll
    for (int e = 0; e < 8; ++e) comb[(size_t)t * 8 + e] = 0.f;
    comb[(size_t)t * 8 + i1] = inv;
    comb[(size_t)t * 8 + i2] = e2 * inv;
}

__global__ __launch_bounds__(256) void k_copy(const float* __restrict__ src,
                                              float* __restrict__ dst, size_t n) {
    size_t i = (size_t)blockIdx.x * 256 + threadIdx.x;
    if (i < n) dst[i] = src[i];
}

// ---------------------------------------------------------------------------
// Host orchestration
// ---------------------------------------------------------------------------
extern "C" void kernel_launch(void* const* d_in, const int* in_sizes, int n_in,
                              void* d_out, int out_size, void* d_ws, size_t ws_size,
                              hipStream_t stream) {
    (void)in_sizes; (void)n_in; (void)out_size; (void)ws_size;
    const int Bb = 4, S = 1024, D = 1024, NH = 16, F = 2048, NE = 8;
    const size_t NT = (size_t)Bb * S;  // 4096 tokens

    const float* x     = (const float*)d_in[0];
    const int*   mask  = (const int*)d_in[1];
    const float* ln1g  = (const float*)d_in[2];
    const float* ln1bb = (const float*)d_in[3];
    const float* Wd[8] = {(const float*)d_in[4],  (const float*)d_in[5],
                          (const float*)d_in[6],  (const float*)d_in[7],
                          (const float*)d_in[8],  (const float*)d_in[9],
                          (const float*)d_in[10], (const float*)d_in[11]};
    const float* wgate = (const float*)d_in[12];
    const float* ln2g  = (const float*)d_in[13];
    const float* ln2bb = (const float*)d_in[14];
    const float* wrout = (const float*)d_in[15];
    const float* ew1   = (const float*)d_in[16];
    const float* ew2   = (const float*)d_in[17];
    float* outF = (float*)d_out;

    char* wp = (char*)d_ws;
    auto alloc = [&](size_t bytes) -> char* {
        char* p = wp;
        wp += (bytes + 255) & ~(size_t)255;
        return p;
    };
    float* xn1 = (float*)alloc(NT * D * 4);
    unsigned short* xn1b = (unsigned short*)alloc(NT * D * 2);
    float* proj[6];  // Q K V Qd Kd Vd
    for (int i = 0; i < 6; ++i) proj[i] = (float*)alloc(NT * D * 4);
    unsigned short* wT[8];
    for (int i = 0; i < 8; ++i) wT[i] = (unsigned short*)alloc((size_t)D * D * 2);
    unsigned short* Qb = (unsigned short*)alloc(NT * D * 2);
    unsigned short* Kb = (unsigned short*)alloc(NT * D * 2);
    unsigned short* Vb = (unsigned short*)alloc(NT * D * 2);
    float* attn = (float*)alloc(NT * D * 4);
    unsigned short* attnb = (unsigned short*)alloc(NT * D * 2);
    float* ctx  = (float*)alloc((size_t)Bb * NH * 64 * 64 * 4);
    float* ksum = (float*)alloc((size_t)Bb * NH * 64 * 4);
    float* dpre = (float*)alloc(NT * D * 4);
    unsigned short* dpreb = (unsigned short*)alloc(NT * D * 2);
    float* dense = (float*)alloc(NT * D * 4);
    float* delta = (float*)alloc(NT * D * 4);
    float* g0 = (float*)alloc(NT * 4);
    float* x2 = (float*)alloc(NT * D * 4);
    float* xn2 = (float*)alloc(NT * D * 4);
    unsigned short* xn2b = (unsigned short*)alloc(NT * D * 2);
    float* comb = (float*)alloc(NT * 8 * 4);
    unsigned short* ew1T = (unsigned short*)alloc((size_t)F * D * 2);
    unsigned short* ew2T = (unsigned short*)alloc((size_t)D * F * 2);
    unsigned short* Hb = (unsigned short*)alloc(NT * F * 2);

    const int cvtBlocks = (int)((NT * D + 255) / 256);
    const dim3 gq(D / 128, (unsigned)(NT / 128));  // projections: N=1024

    // 1) LN1
    k_layernorm<<<(unsigned)NT, 256, 0, stream>>>(x, ln1g, ln1bb, xn1, xn1b);
    // 2) weight transpose-converts (B^T staging)
    {
        int blocks = (int)(((size_t)D * D + 255) / 256);
        for (int i = 0; i < 8; ++i)
            k_tcvt<<<blocks, 256, 0, stream>>>(Wd[i], wT[i], D, D);
    }
    // 3) six projections: Q,K,V from wq,wk,wv; Qd,Kd,Vd from wqd,wkd,wvd
    const int order[6] = {0, 1, 2, 4, 5, 6};
    for (int i = 0; i < 6; ++i)
        k_gemm<<<gq, 256, 0, stream>>>(xn1b, wT[order[i]], proj[i], nullptr, nullptr,
                                       0, 0, D, D, 0);
    // 4) bf16 copies of Q,K,V for the attention WMMA kernel
    k_cvt<<<cvtBlocks, 256, 0, stream>>>(proj[0], Qb, NT * D);
    k_cvt<<<cvtBlocks, 256, 0, stream>>>(proj[1], Kb, NT * D);
    k_cvt<<<cvtBlocks, 256, 0, stream>>>(proj[2], Vb, NT * D);
    // 5) flash attention (dense branch)
    k_flash<<<dim3(S / 64, NH, Bb), 128, 0, stream>>>(Qb, Kb, Vb, mask, attn);
    // 6) dense = attn @ wo
    k_cvt<<<cvtBlocks, 256, 0, stream>>>(attn, attnb, NT * D);
    k_gemm<<<gq, 256, 0, stream>>>(attnb, wT[3], dense, nullptr, nullptr, 0, 0, D, D, 0);
    // 7) linear delta branch
    k_linctx<<<dim3(NH, Bb), 256, 0, stream>>>(proj[4], proj[5], ctx, ksum);
    k_linout<<<dim3((unsigned)NT, NH), 64, 0, stream>>>(proj[3], ctx, ksum, dpre);
    k_cvt<<<cvtBlocks, 256, 0, stream>>>(dpre, dpreb, NT * D);
    k_gemm<<<gq, 256, 0, stream>>>(dpreb, wT[7], delta, nullptr, nullptr, 0, 0, D, D, 0);
    // 8) gate + residual combine
    k_gate<<<(unsigned)((NT + 255) / 256), 256, 0, stream>>>(xn1, wgate, g0, (int)NT);
    k_combine<<<cvtBlocks, 256, 0, stream>>>(x, dense, delta, g0, x2, D, NT * D);
    // 9) LN2 + router
    k_layernorm<<<(unsigned)NT, 256, 0, stream>>>(x2, ln2g, ln2bb, xn2, xn2b);
    k_router<<<(unsigned)((NT + 255) / 256), 256, 0, stream>>>(xn2, wrout, comb, (int)NT);
    // 10) out = x2 (residual), then experts accumulate into it
    k_copy<<<cvtBlocks, 256, 0, stream>>>(x2, outF, NT * D);
    // 11) MoE: per expert, GEMM1 (GELU->bf16 fused) then GEMM2 (scaled accumulate)
    {
        const dim3 g1(F / 128, (unsigned)(NT / 128));
        const dim3 g2(D / 128, (unsigned)(NT / 128));
        const int tb = (int)(((size_t)D * F + 255) / 256);
        for (int e = 0; e < NE; ++e) {
            k_tcvt<<<tb, 256, 0, stream>>>(ew1 + (size_t)e * D * F, ew1T, D, F);
            k_tcvt<<<tb, 256, 0, stream>>>(ew2 + (size_t)e * F * D, ew2T, F, D);
            k_gemm<<<g1, 256, 0, stream>>>(xn2b, ew1T, nullptr, Hb, nullptr, 0, 0,
                                           F, D, 1);
            k_gemm<<<g2, 256, 0, stream>>>(Hb, ew2T, outF, nullptr, comb, 8, e,
                                           D, F, 2);
        }
    }
}
